// MultiHeadAttention_11149735100694
// MI455X (gfx1250) — matrix operand order repaired and checked by reading, not compiled, not run on hardware
//
#include <hip/hip_runtime.h>
#include <hip/hip_bf16.h>

// ---------------------------------------------------------------------------
// MHA forward for B=2, S=2048, D=1024, H=16, HD=64 on gfx1250 (wave32, WMMA)
// ---------------------------------------------------------------------------

typedef __attribute__((ext_vector_type(16))) _Float16 v16h;
typedef __attribute__((ext_vector_type(8)))  _Float16 v8h;
typedef __attribute__((ext_vector_type(4)))  _Float16 v4h;
typedef __attribute__((ext_vector_type(8)))  float    v8f;
typedef __attribute__((ext_vector_type(4)))  float    v4f;

#define S_LEN  2048
#define D_DIM  1024
#define H_NUM  16
#define HD_DIM 64

// ---------------- fragment loaders -----------------------------------------

// A-matrix 16x32 f16 fragment from row-major [*, lda] at (row0, k0).
// Lane L: row = L&15, two 8-half chunks at K = k0 + 8*(L>>4) and +16.
__device__ __forceinline__ v16h load_a_f16(const _Float16* __restrict__ base,
                                           int lda, int row0, int k0, int lane) {
  int r = lane & 15;
  int koff = (lane >> 4) << 3;
  const _Float16* p = base + (size_t)(row0 + r) * lda + k0 + koff;
  v8h lo = *(const v8h*)p;
  v8h hi = *(const v8h*)(p + 16);
  v16h out;
#pragma unroll
  for (int i = 0; i < 8; ++i) { out[i] = lo[i]; out[i + 8] = hi[i]; }
  return out;
}

// Same A fragment but source is f32; convert in-register.
__device__ __forceinline__ v16h load_a_f32(const float* __restrict__ base,
                                           int lda, int row0, int k0, int lane) {
  int r = lane & 15;
  int koff = (lane >> 4) << 3;
  const float* p = base + (size_t)(row0 + r) * lda + k0 + koff;
  v4f a0 = *(const v4f*)p;
  v4f a1 = *(const v4f*)(p + 4);
  v4f b0 = *(const v4f*)(p + 16);
  v4f b1 = *(const v4f*)(p + 20);
  v16h out;
#pragma unroll
  for (int i = 0; i < 4; ++i) {
    out[i]      = (_Float16)a0[i];
    out[i + 4]  = (_Float16)a1[i];
    out[i + 8]  = (_Float16)b0[i];
    out[i + 12] = (_Float16)b1[i];
  }
  return out;
}

// B-matrix 32x16 fragment where the K-dim is contiguous in memory:
// source is row-major [N, ldb] (e.g. torch weight [out,in]); B = W^T.
// Lane L: col = L&15, two 8-half chunks at K = k0 + 8*(L>>4) and +16.
__device__ __forceinline__ v16h load_b_f16(const _Float16* __restrict__ base,
                                           int ldb, int n0, int k0, int lane) {
  int n = lane & 15;
  int koff = (lane >> 4) << 3;
  const _Float16* p = base + (size_t)(n0 + n) * ldb + k0 + koff;
  v8h lo = *(const v8h*)p;
  v8h hi = *(const v8h*)(p + 16);
  v16h out;
#pragma unroll
  for (int i = 0; i < 8; ++i) { out[i] = lo[i]; out[i + 8] = hi[i]; }
  return out;
}

__device__ __forceinline__ v8f wmma_f16(v16h a, v16h b, v8f c) {
  return __builtin_amdgcn_wmma_f32_16x16x32_f16(
      /*neg_a=*/false, a, /*neg_b=*/false, b,
      /*c_mod=*/(short)0, c, /*reuse_a=*/false, /*reuse_b=*/false);
}

// ---------------- kernels ----------------------------------------------------

// f32 -> f16, 4 elements per thread (n multiple of 4)
__global__ void cvt_f32_to_f16_kernel(const float* __restrict__ in,
                                      _Float16* __restrict__ out, int n) {
  int i = (blockIdx.x * blockDim.x + threadIdx.x) * 4;
  if (i < n) {
    v4f v = *(const v4f*)(in + i);
    v4h h;
#pragma unroll
    for (int j = 0; j < 4; ++j) h[j] = (_Float16)v[j];
    *(v4h*)(out + i) = h;
  }
}

// C = A[M,K] @ W[N,K]^T + bias.  4 waves (2x2), 64x64 per wave, 128x128/block.
// mode 0: store f16 head-major Q/K  [B,H,S,HD]
// mode 1: store f16 transposed V    [B,H,HD,S]
// mode 2: store f32 row-major       [M,N]     (final output projection)
__global__ __launch_bounds__(128)
void gemm_proj_kernel(const _Float16* __restrict__ A,
                      const _Float16* __restrict__ W,
                      const float* __restrict__ bias,
                      _Float16* __restrict__ out_h,
                      float* __restrict__ out_f,
                      int Kdim, int Ndim, int mode) {
  int lane = threadIdx.x & 31;
  int wave = threadIdx.x >> 5;
  int row0 = blockIdx.x * 128 + (wave >> 1) * 64;
  int col0 = blockIdx.y * 128 + (wave & 1) * 64;

  v8f zero = {};
  v8f acc[4][4];
#pragma unroll
  for (int mi = 0; mi < 4; ++mi)
#pragma unroll
    for (int ni = 0; ni < 4; ++ni) acc[mi][ni] = zero;

  for (int kk = 0; kk < Kdim; kk += 32) {
    v16h af[4], bf[4];
#pragma unroll
    for (int mi = 0; mi < 4; ++mi)
      af[mi] = load_a_f16(A, Kdim, row0 + mi * 16, kk, lane);
#pragma unroll
    for (int ni = 0; ni < 4; ++ni)
      bf[ni] = load_b_f16(W, Kdim, col0 + ni * 16, kk, lane);
#pragma unroll
    for (int mi = 0; mi < 4; ++mi)
#pragma unroll
      for (int ni = 0; ni < 4; ++ni)
        acc[mi][ni] = wmma_f16(af[mi], bf[ni], acc[mi][ni]);
  }

  int rsel = (lane >> 4) * 8;
  int ncol = lane & 15;
#pragma unroll
  for (int mi = 0; mi < 4; ++mi)
#pragma unroll
    for (int ni = 0; ni < 4; ++ni)
#pragma unroll
      for (int i = 0; i < 8; ++i) {
        int m = row0 + mi * 16 + rsel + i;
        int n = col0 + ni * 16 + ncol;
        float v = acc[mi][ni][i] + bias[n];
        if (mode == 0) {
          int b = m >> 11, s = m & (S_LEN - 1), h = n >> 6, hd = n & (HD_DIM - 1);
          out_h[(((size_t)(b * H_NUM + h) * S_LEN) + s) * HD_DIM + hd] = (_Float16)v;
        } else if (mode == 1) {
          int b = m >> 11, s = m & (S_LEN - 1), h = n >> 6, hd = n & (HD_DIM - 1);
          out_h[(((size_t)(b * H_NUM + h) * HD_DIM) + hd) * S_LEN + s] = (_Float16)v;
        } else {
          out_f[(size_t)m * Ndim + n] = v;
        }
      }
}

// energy[bh,q,k] = (Q . K) / 8, causal blocks only (upper-right blocks skipped)
__global__ __launch_bounds__(128)
void attn_qk_kernel(const _Float16* __restrict__ Qh,
                    const _Float16* __restrict__ Kh,
                    float* __restrict__ energy) {
  int bh = blockIdx.z;
  int q0 = blockIdx.x * 128;
  int k0 = blockIdx.y * 128;
  if (k0 > q0 + 127) return;  // fully masked 128x128 block

  const _Float16* Qp = Qh + (size_t)bh * S_LEN * HD_DIM;
  const _Float16* Kp = Kh + (size_t)bh * S_LEN * HD_DIM;
  float* Ep = energy + (size_t)bh * S_LEN * S_LEN;

  int lane = threadIdx.x & 31;
  int wave = threadIdx.x >> 5;
  int row0 = q0 + (wave >> 1) * 64;
  int col0 = k0 + (wave & 1) * 64;
  if (col0 > row0 + 63) return;  // fully masked 64x64 wave tile

  v8f zero = {};
  v8f acc[4][4];
#pragma unroll
  for (int mi = 0; mi < 4; ++mi)
#pragma unroll
    for (int ni = 0; ni < 4; ++ni) acc[mi][ni] = zero;

#pragma unroll
  for (int kk = 0; kk < HD_DIM; kk += 32) {
    v16h af[4], bf[4];
#pragma unroll
    for (int mi = 0; mi < 4; ++mi)
      af[mi] = load_a_f16(Qp, HD_DIM, row0 + mi * 16, kk, lane);
#pragma unroll
    for (int ni = 0; ni < 4; ++ni)
      bf[ni] = load_b_f16(Kp, HD_DIM, col0 + ni * 16, kk, lane);
#pragma unroll
    for (int mi = 0; mi < 4; ++mi)
#pragma unroll
      for (int ni = 0; ni < 4; ++ni)
        acc[mi][ni] = wmma_f16(af[mi], bf[ni], acc[mi][ni]);
  }

  const float scale = 0.125f;  // 1/sqrt(64)
  int rsel = (lane >> 4) * 8;
  int ncol = lane & 15;
#pragma unroll
  for (int mi = 0; mi < 4; ++mi)
#pragma unroll
    for (int ni = 0; ni < 4; ++ni)
#pragma unroll
      for (int i = 0; i < 8; ++i) {
        int m = row0 + mi * 16 + rsel + i;
        int n = col0 + ni * 16 + ncol;
        Ep[(size_t)m * S_LEN + n] = acc[mi][ni][i] * scale;
      }
}

// In-place causal row softmax over attn rows of length 2048.
__global__ __launch_bounds__(256)
void softmax_kernel(float* __restrict__ attn) {
  __shared__ float red[256];
  int row = blockIdx.x;          // bh*S + q
  int q = row & (S_LEN - 1);
  int nvalid = q + 1;
  float* p = attn + (size_t)row * S_LEN;
  int tid = threadIdx.x;

  float vals[8];
  float mx = -3.4e38f;
#pragma unroll
  for (int j = 0; j < 8; ++j) {
    int idx = tid + j * 256;
    float v = (idx < nvalid) ? p[idx] : -3.4e38f;
    vals[j] = v;
    mx = fmaxf(mx, v);
  }
  red[tid] = mx;
  __syncthreads();
  for (int s = 128; s > 0; s >>= 1) {
    if (tid < s) red[tid] = fmaxf(red[tid], red[tid + s]);
    __syncthreads();
  }
  mx = red[0];
  __syncthreads();

  float sum = 0.0f;
#pragma unroll
  for (int j = 0; j < 8; ++j) {
    int idx = tid + j * 256;
    if (idx < nvalid) {
      float e = __expf(vals[j] - mx);
      vals[j] = e;
      sum += e;
    }
  }
  red[tid] = sum;
  __syncthreads();
  for (int s = 128; s > 0; s >>= 1) {
    if (tid < s) red[tid] += red[tid + s];
    __syncthreads();
  }
  float inv = 1.0f / red[0];

#pragma unroll
  for (int j = 0; j < 8; ++j) {
    int idx = tid + j * 256;
    p[idx] = (idx < nvalid) ? vals[j] * inv : 0.0f;
  }
}

// ctx[b, q, h*64+d] = sum_k P[bh,q,k] * V[bh,k,d]; V supplied transposed [bh,d,k].
// 4 waves stacked in M: 256x64 block tile; causal K bound.
__global__ __launch_bounds__(128)
void attn_pv_kernel(const float* __restrict__ attn,
                    const _Float16* __restrict__ Vt,
                    _Float16* __restrict__ ctx) {
  int bh = blockIdx.z;
  int b = bh >> 4, h = bh & (H_NUM - 1);
  int lane = threadIdx.x & 31;
  int wave = threadIdx.x >> 5;
  int row0 = blockIdx.x * 256 + wave * 64;

  const float* P = attn + (size_t)bh * S_LEN * S_LEN;
  const _Float16* V = Vt + (size_t)bh * HD_DIM * S_LEN;

  v8f zero = {};
  v8f acc[4][4];
#pragma unroll
  for (int mi = 0; mi < 4; ++mi)
#pragma unroll
    for (int ni = 0; ni < 4; ++ni) acc[mi][ni] = zero;

  int klim = row0 + 64;  // rows [row0, row0+63] only attend to k <= row0+63
  for (int kk = 0; kk < klim; kk += 32) {
    v16h af[4], bf[4];
#pragma unroll
    for (int mi = 0; mi < 4; ++mi)
      af[mi] = load_a_f32(P, S_LEN, row0 + mi * 16, kk, lane);
#pragma unroll
    for (int ni = 0; ni < 4; ++ni)
      bf[ni] = load_b_f16(V, S_LEN, ni * 16, kk, lane);
#pragma unroll
    for (int mi = 0; mi < 4; ++mi)
#pragma unroll
      for (int ni = 0; ni < 4; ++ni)
        acc[mi][ni] = wmma_f16(af[mi], bf[ni], acc[mi][ni]);
  }

  int rsel = (lane >> 4) * 8;
  int ncol = lane & 15;
#pragma unroll
  for (int mi = 0; mi < 4; ++mi)
#pragma unroll
    for (int ni = 0; ni < 4; ++ni)
#pragma unroll
      for (int i = 0; i < 8; ++i) {
        int m = row0 + mi * 16 + rsel + i;   // q
        int n = ni * 16 + ncol;              // d in [0,64)
        ctx[((size_t)(b * S_LEN + m) * D_DIM) + h * HD_DIM + n] =
            (_Float16)acc[mi][ni][i];
      }
}

// ---------------- host launcher ---------------------------------------------

extern "C" void kernel_launch(void* const* d_in, const int* in_sizes, int n_in,
                              void* d_out, int out_size, void* d_ws, size_t ws_size,
                              hipStream_t stream) {
  (void)in_sizes; (void)n_in; (void)out_size; (void)ws_size;

  const float* query = (const float*)d_in[0];
  const float* key_t = (const float*)d_in[1];
  const float* value = (const float*)d_in[2];
  // d_in[3] = mask (causal; implemented analytically)
  const float* Wq = (const float*)d_in[4];  const float* bq = (const float*)d_in[5];
  const float* Wk = (const float*)d_in[6];  const float* bk = (const float*)d_in[7];
  const float* Wv = (const float*)d_in[8];  const float* bv = (const float*)d_in[9];
  const float* Wo = (const float*)d_in[10]; const float* bo = (const float*)d_in[11];

  const size_t BSD = (size_t)2 * S_LEN * D_DIM;   // 4,194,304
  const size_t DD  = (size_t)D_DIM * D_DIM;       // 1,048,576

  _Float16* ws = (_Float16*)d_ws;
  _Float16* Xq  = ws;                 // converted inputs
  _Float16* Xk  = Xq + BSD;
  _Float16* Xv  = Xk + BSD;
  _Float16* Wqh = Xv + BSD;           // converted weights
  _Float16* Wkh = Wqh + DD;
  _Float16* Wvh = Wkh + DD;
  _Float16* Woh = Wvh + DD;
  _Float16* Qh  = Woh + DD;           // [B,H,S,HD] f16
  _Float16* Kh  = Qh + BSD;           // [B,H,S,HD] f16
  _Float16* Vth = Kh + BSD;           // [B,H,HD,S] f16
  _Float16* Ctx = Vth + BSD;          // [B,S,D]    f16

  float* out_x = (float*)d_out;                   // [B,S,D] f32
  float* attn  = out_x + BSD;                     // [B,H,S,S] f32

  // 1) f32 -> f16 conversions
  cvt_f32_to_f16_kernel<<<BSD / 4 / 256, 256, 0, stream>>>(query, Xq, (int)BSD);
  cvt_f32_to_f16_kernel<<<BSD / 4 / 256, 256, 0, stream>>>(key_t, Xk, (int)BSD);
  cvt_f32_to_f16_kernel<<<BSD / 4 / 256, 256, 0, stream>>>(value, Xv, (int)BSD);
  cvt_f32_to_f16_kernel<<<DD / 4 / 256, 256, 0, stream>>>(Wq, Wqh, (int)DD);
  cvt_f32_to_f16_kernel<<<DD / 4 / 256, 256, 0, stream>>>(Wk, Wkh, (int)DD);
  cvt_f32_to_f16_kernel<<<DD / 4 / 256, 256, 0, stream>>>(Wv, Wvh, (int)DD);
  cvt_f32_to_f16_kernel<<<DD / 4 / 256, 256, 0, stream>>>(Wo, Woh, (int)DD);

  // 2) Q/K/V projections (M=4096, N=1024, K=1024)
  dim3 pgrid(4096 / 128, 1024 / 128, 1);
  gemm_proj_kernel<<<pgrid, 128, 0, stream>>>(Xq, Wqh, bq, Qh, nullptr, D_DIM, D_DIM, 0);
  gemm_proj_kernel<<<pgrid, 128, 0, stream>>>(Xk, Wkh, bk, Kh, nullptr, D_DIM, D_DIM, 0);
  gemm_proj_kernel<<<pgrid, 128, 0, stream>>>(Xv, Wvh, bv, Vth, nullptr, D_DIM, D_DIM, 1);

  // 3) energy = Q K^T / 8 (causal blocks only)
  dim3 qkgrid(S_LEN / 128, S_LEN / 128, 2 * H_NUM);
  attn_qk_kernel<<<qkgrid, 128, 0, stream>>>(Qh, Kh, attn);

  // 4) causal softmax in place -> attention output
  softmax_kernel<<<2 * H_NUM * S_LEN, 256, 0, stream>>>(attn);

  // 5) ctx = attention @ V
  dim3 pvgrid(S_LEN / 256, 1, 2 * H_NUM);
  attn_pv_kernel<<<pvgrid, 128, 0, stream>>>(attn, Vth, Ctx);

  // 6) x = ctx @ Wo^T + bo  (f32 output)
  gemm_proj_kernel<<<pgrid, 128, 0, stream>>>(Ctx, Woh, bo, nullptr, out_x, D_DIM, D_DIM, 2);
}
